// DismantledBlock_52097953300547
// MI455X (gfx1250) — compile-verified, matrix operand order checked by reading, not run on hardware
//
#include <hip/hip_runtime.h>
#include <hip/hip_bf16.h>
#include <stdint.h>

typedef __attribute__((ext_vector_type(16))) __bf16        bf16x16;
typedef __attribute__((ext_vector_type(8)))  float         f32x8;
typedef __attribute__((ext_vector_type(4)))  unsigned int  uint4v;
typedef int int4v __attribute__((vector_size(16)));

#define DEV __device__ __forceinline__

#if __has_builtin(__builtin_amdgcn_global_load_async_to_lds_b128) && \
    __has_builtin(__builtin_amdgcn_s_wait_asynccnt)
#define USE_ASYNC_LDS 1
#define AS1 __attribute__((address_space(1)))
#define AS3 __attribute__((address_space(3)))
#else
#define USE_ASYNC_LDS 0
#endif

static constexpr int Bn = 2, Tn = 2048, Dn = 1024, Hn = 16, DhN = 64, HmN = 2816;
static constexpr int Mn = Bn * Tn;          // 4096 token rows
static constexpr float RMS_EPS = 1e-6f;

DEV unsigned short f2bf(float f) {
  unsigned int u = __builtin_bit_cast(unsigned int, f);
  u += 0x7fffu + ((u >> 16) & 1u);          // round-to-nearest-even
  return (unsigned short)(u >> 16);
}
DEV float siluf(float x) { return x / (1.0f + __expf(-x)); }
DEV f32x8 zero8() {
  f32x8 z;
#pragma unroll
  for (int i = 0; i < 8; ++i) z[i] = 0.0f;
  return z;
}
// Build a 16x32-bf16 WMMA fragment from two contiguous 16-byte runs.
DEV bf16x16 mk_frag(const unsigned short* lo, const unsigned short* hi) {
  union { uint4v u[2]; bf16x16 f; } un;
  un.u[0] = *(const uint4v*)lo;
  un.u[1] = *(const uint4v*)hi;
  return un.f;
}
// byte shuffles for 2x2 bf16 transpose: (a.lo,b.lo) and (a.hi,b.hi)
DEV unsigned int lo_pair(unsigned int a, unsigned int b) {
  return __builtin_amdgcn_perm(b, a, 0x05040100u);
}
DEV unsigned int hi_pair(unsigned int a, unsigned int b) {
  return __builtin_amdgcn_perm(b, a, 0x07060302u);
}
// global->LDS 16B copy: async (ASYNCcnt) when available, else through VGPRs
DEV void cp16(const unsigned short* g, unsigned short* l) {
#if USE_ASYNC_LDS
  __builtin_amdgcn_global_load_async_to_lds_b128(
      (AS1 int4v*)(uintptr_t)g,
      (AS3 int4v*)(unsigned int)(uintptr_t)l,
      0, 0);
#else
  *(uint4v*)l = *(const uint4v*)g;
#endif
}
DEV void cp16_commit() {
#if USE_ASYNC_LDS
  __builtin_amdgcn_s_wait_asynccnt(0);
#endif
}

// ---------------------------------------------------------------------------
// fp32 -> bf16 flat conversion (weights); 4 elems/thread, packed u32 stores
// ---------------------------------------------------------------------------
__global__ void k_cvt(const float* __restrict__ src, unsigned short* __restrict__ dst, int n) {
  int i = (blockIdx.x * 256 + threadIdx.x) * 4;
  if (i < n) {
    unsigned int p0 = (unsigned int)f2bf(src[i]) | ((unsigned int)f2bf(src[i + 1]) << 16);
    unsigned int p1 = (unsigned int)f2bf(src[i + 2]) | ((unsigned int)f2bf(src[i + 3]) << 16);
    *(unsigned int*)&dst[i] = p0;
    *(unsigned int*)&dst[i + 2] = p1;
  }
}

// ---------------------------------------------------------------------------
// csilu = silu(c)  (tiny: B*D elements)
// ---------------------------------------------------------------------------
__global__ void k_silu(const float* __restrict__ c, float* __restrict__ o, int n) {
  int i = blockIdx.x * 256 + threadIdx.x;
  if (i < n) o[i] = siluf(c[i]);
}

// ---------------------------------------------------------------------------
// mods = csilu @ adaln_w + adaln_b   ([B,6D], tiny)
// ---------------------------------------------------------------------------
__global__ __launch_bounds__(256) void k_adaln(const float* __restrict__ cs,
                                               const float* __restrict__ w,
                                               const float* __restrict__ bias,
                                               float* __restrict__ mods) {
  int j = blockIdx.x * 256 + threadIdx.x;       // 0 .. 2*6144-1
  int b = j / (6 * Dn);
  int col = j - b * (6 * Dn);
  const float* crow = cs + b * Dn;
  float acc = 0.0f;
#pragma unroll 4
  for (int k = 0; k < Dn; ++k) acc += crow[k] * w[(size_t)k * (6 * Dn) + col];
  mods[j] = acc + bias[col];
}

// ---------------------------------------------------------------------------
// x_mod = RMS(x) * (1+scale_msa) + shift_msa  -> bf16     (one block per row)
// ---------------------------------------------------------------------------
__global__ __launch_bounds__(256) void k_modrms(const float* __restrict__ x,
                                                const float* __restrict__ mods,
                                                unsigned short* __restrict__ xmod) {
  int row = blockIdx.x;
  int b = row >> 11;                              // T = 2048
  int tid = threadIdx.x;
  const float* xr = x + (size_t)row * Dn;
  float v[4];
  float ss = 0.0f;
#pragma unroll
  for (int j = 0; j < 4; ++j) { v[j] = xr[tid * 4 + j]; ss += v[j] * v[j]; }
  __shared__ float red[256];
  red[tid] = ss;
  __syncthreads();
  for (int s = 128; s > 0; s >>= 1) {
    if (tid < s) red[tid] += red[tid + s];
    __syncthreads();
  }
  float rs = rsqrtf(red[0] * (1.0f / Dn) + RMS_EPS);
  const float* mo = mods + b * (6 * Dn);
  unsigned int pk[2];
#pragma unroll
  for (int j = 0; j < 4; ++j) {
    int d = tid * 4 + j;
    float y = v[j] * rs * (1.0f + mo[Dn + d]) + mo[d];   // scale_msa, shift_msa
    unsigned short h = f2bf(y);
    if (j & 1) pk[j >> 1] |= ((unsigned int)h << 16);
    else       pk[j >> 1]  = h;
  }
  *(unsigned int*)&xmod[(size_t)row * Dn + tid * 4]     = pk[0];
  *(unsigned int*)&xmod[(size_t)row * Dn + tid * 4 + 2] = pk[1];
}

// ---------------------------------------------------------------------------
// bf16 WMMA GEMM:  C[M,N](f32) = A[M,K](bf16) @ B[K,N](bf16) (+ bias[n])
// N,K compile-time so strides fold into immediate offsets.
// 128x128x32 tiles, 256 threads = 8 waves (2x4 wave grid, 64x32 per wave)
// A tile async-copied to LDS; B tile staged TRANSPOSED via v_perm shuffles.
// ---------------------------------------------------------------------------
template <int N, int K>
__global__ __launch_bounds__(256) void k_gemm(const unsigned short* __restrict__ A,
                                              const unsigned short* __restrict__ Bw,
                                              float* __restrict__ C,
                                              const float* __restrict__ bias) {
  __shared__ unsigned short sA[128 * 32];
  __shared__ unsigned short sBt[128 * 32];      // [n][k] transposed
  int tid = threadIdx.x, lane = tid & 31, wid = tid >> 5;
  int wm = wid >> 2, wn = wid & 3;
  int m0 = blockIdx.y * 128, n0 = blockIdx.x * 128;
  int ml = lane & 15, kb = (lane >> 4) * 8, mb = (lane < 16) ? 0 : 8;
  int np = tid & 63, kq = tid >> 6;             // B-stage: n-pair 0..63, k-octet 0..3

  f32x8 acc[4][2];
#pragma unroll
  for (int mt = 0; mt < 4; ++mt)
#pragma unroll
    for (int nt = 0; nt < 2; ++nt) acc[mt][nt] = zero8();

  for (int k0 = 0; k0 < K; k0 += 32) {
    const unsigned short* gA = A + (size_t)m0 * K + k0;
    const unsigned short* gB = Bw + (size_t)k0 * N + n0;
    // A tile: 128x32 bf16, async b128 copies
#pragma unroll
    for (int j = 0; j < 2; ++j) {
      int idx = tid + j * 256;                  // 512 x uint4
      int r = idx >> 2, cc = idx & 3;
      cp16(&gA[(size_t)r * K + cc * 8], &sA[r * 32 + cc * 8]);
    }
    // B tile: 32x128 bf16 -> transposed [n][k] via v_perm pair shuffles
    {
      unsigned int in[8];
#pragma unroll
      for (int i = 0; i < 8; ++i)
        in[i] = *(const unsigned int*)&gB[(size_t)(kq * 8 + i) * N + np * 2];
      uint4v ta, tb;
#pragma unroll
      for (int j2 = 0; j2 < 4; ++j2) {
        ta[j2] = lo_pair(in[2 * j2], in[2 * j2 + 1]);
        tb[j2] = hi_pair(in[2 * j2], in[2 * j2 + 1]);
      }
      *(uint4v*)&sBt[(np * 2 + 0) * 32 + kq * 8] = ta;
      *(uint4v*)&sBt[(np * 2 + 1) * 32 + kq * 8] = tb;
    }
    if (k0 + 32 < K) {
      __builtin_prefetch(&gA[(size_t)(tid >> 2) * K + (tid & 3) * 8 + 32], 0, 0);
      __builtin_prefetch(&gB[(size_t)(kq * 8) * N + 32 * (size_t)N + np * 2], 0, 0);
    }
    cp16_commit();
    __syncthreads();

    bf16x16 af[4], bfr[2];
#pragma unroll
    for (int mt = 0; mt < 4; ++mt) {
      int r = wm * 64 + mt * 16 + ml;
      af[mt] = mk_frag(&sA[r * 32 + kb], &sA[r * 32 + kb + 16]);
    }
#pragma unroll
    for (int nt = 0; nt < 2; ++nt) {
      int cc = wn * 32 + nt * 16 + ml;
      bfr[nt] = mk_frag(&sBt[cc * 32 + kb], &sBt[cc * 32 + kb + 16]);
    }
#pragma unroll
    for (int mt = 0; mt < 4; ++mt)
#pragma unroll
      for (int nt = 0; nt < 2; ++nt)
        acc[mt][nt] = __builtin_amdgcn_wmma_f32_16x16x32_bf16(
            false, af[mt], false, bfr[nt], (short)0, acc[mt][nt], false, false);
    __syncthreads();
  }

#pragma unroll
  for (int mt = 0; mt < 4; ++mt)
#pragma unroll
    for (int nt = 0; nt < 2; ++nt) {
      int cc = n0 + wn * 32 + nt * 16 + ml;
      float bv = bias ? bias[cc] : 0.0f;
#pragma unroll
      for (int v = 0; v < 8; ++v) {
        int r = m0 + wm * 64 + mt * 16 + mb + v;
        C[(size_t)r * N + cc] = acc[mt][nt][v] + bv;
      }
    }
}

// ---------------------------------------------------------------------------
// Dual-B WMMA GEMM for SwiGLU: G = bf16( silu(A@W1) * (A@W3) )
// 64x64x32 tiles, 128 threads = 4 waves (2x2); B tiles staged transposed.
// ---------------------------------------------------------------------------
template <int N, int K>
__global__ __launch_bounds__(128) void k_gemm_dual(const unsigned short* __restrict__ A,
                                                   const unsigned short* __restrict__ B1,
                                                   const unsigned short* __restrict__ B3,
                                                   unsigned short* __restrict__ G) {
  __shared__ unsigned short sA[64 * 32];
  __shared__ unsigned short sB1t[64 * 32];
  __shared__ unsigned short sB3t[64 * 32];
  int tid = threadIdx.x, lane = tid & 31, wid = tid >> 5;
  int wm = wid >> 1, wn = wid & 1;
  int m0 = blockIdx.y * 64, n0 = blockIdx.x * 64;
  int ml = lane & 15, kb = (lane >> 4) * 8, mb = (lane < 16) ? 0 : 8;
  int np = tid & 31, kq = tid >> 5;             // n-pair 0..31, k-octet 0..3

  f32x8 a1[2][2], a3[2][2];
#pragma unroll
  for (int mt = 0; mt < 2; ++mt)
#pragma unroll
    for (int nt = 0; nt < 2; ++nt) { a1[mt][nt] = zero8(); a3[mt][nt] = zero8(); }

  for (int k0 = 0; k0 < K; k0 += 32) {
    const unsigned short* gA = A + (size_t)m0 * K + k0;
    const unsigned short* gB1 = B1 + (size_t)k0 * N + n0;
    const unsigned short* gB3 = B3 + (size_t)k0 * N + n0;
#pragma unroll
    for (int j = 0; j < 2; ++j) {
      int idx = tid + j * 128;                  // 256 x uint4
      int r = idx >> 2, cc = idx & 3;
      cp16(&gA[(size_t)r * K + cc * 8], &sA[r * 32 + cc * 8]);
    }
    {
      unsigned int i1[8], i3[8];
#pragma unroll
      for (int i = 0; i < 8; ++i) {
        size_t go = (size_t)(kq * 8 + i) * N + np * 2;
        i1[i] = *(const unsigned int*)&gB1[go];
        i3[i] = *(const unsigned int*)&gB3[go];
      }
      uint4v t1a, t1b, t3a, t3b;
#pragma unroll
      for (int j2 = 0; j2 < 4; ++j2) {
        t1a[j2] = lo_pair(i1[2 * j2], i1[2 * j2 + 1]);
        t1b[j2] = hi_pair(i1[2 * j2], i1[2 * j2 + 1]);
        t3a[j2] = lo_pair(i3[2 * j2], i3[2 * j2 + 1]);
        t3b[j2] = hi_pair(i3[2 * j2], i3[2 * j2 + 1]);
      }
      *(uint4v*)&sB1t[(np * 2 + 0) * 32 + kq * 8] = t1a;
      *(uint4v*)&sB1t[(np * 2 + 1) * 32 + kq * 8] = t1b;
      *(uint4v*)&sB3t[(np * 2 + 0) * 32 + kq * 8] = t3a;
      *(uint4v*)&sB3t[(np * 2 + 1) * 32 + kq * 8] = t3b;
    }
    cp16_commit();
    __syncthreads();

    bf16x16 af[2], b1f[2], b3f[2];
#pragma unroll
    for (int mt = 0; mt < 2; ++mt) {
      int r = wm * 32 + mt * 16 + ml;
      af[mt] = mk_frag(&sA[r * 32 + kb], &sA[r * 32 + kb + 16]);
    }
#pragma unroll
    for (int nt = 0; nt < 2; ++nt) {
      int cc = wn * 32 + nt * 16 + ml;
      b1f[nt] = mk_frag(&sB1t[cc * 32 + kb], &sB1t[cc * 32 + kb + 16]);
      b3f[nt] = mk_frag(&sB3t[cc * 32 + kb], &sB3t[cc * 32 + kb + 16]);
    }
#pragma unroll
    for (int mt = 0; mt < 2; ++mt)
#pragma unroll
      for (int nt = 0; nt < 2; ++nt) {
        a1[mt][nt] = __builtin_amdgcn_wmma_f32_16x16x32_bf16(
            false, af[mt], false, b1f[nt], (short)0, a1[mt][nt], false, false);
        a3[mt][nt] = __builtin_amdgcn_wmma_f32_16x16x32_bf16(
            false, af[mt], false, b3f[nt], (short)0, a3[mt][nt], false, false);
      }
    __syncthreads();
  }

#pragma unroll
  for (int mt = 0; mt < 2; ++mt)
#pragma unroll
    for (int nt = 0; nt < 2; ++nt) {
      int cc = n0 + wn * 32 + nt * 16 + ml;
#pragma unroll
      for (int v = 0; v < 8; ++v) {
        int r = m0 + wm * 32 + mt * 16 + mb + v;
        float g = siluf(a1[mt][nt][v]) * a3[mt][nt][v];
        G[(size_t)r * N + cc] = f2bf(g);
      }
    }
}

// ---------------------------------------------------------------------------
// qk RMS-norm (affine) + RoPE + pack q/k/v to bf16 [B,H,T,Dh]
// one block per (t,b); 16 lanes per head -> wave-half shuffle reductions
// ---------------------------------------------------------------------------
__global__ __launch_bounds__(256) void k_qkvprep(const float* __restrict__ qkv,
                                                 const float* __restrict__ lnq,
                                                 const float* __restrict__ lnk,
                                                 const int* __restrict__ pos,
                                                 unsigned short* __restrict__ qo,
                                                 unsigned short* __restrict__ ko,
                                                 unsigned short* __restrict__ vo) {
  int t = blockIdx.x, b = blockIdx.y;
  int tid = threadIdx.x;
  int head = tid >> 4;                 // 16 threads per head
  int dl0 = (tid & 15) * 4;            // 4 elems each
  const float* row = qkv + (size_t)(b * Tn + t) * (3 * Dn);
  const float* qr = row + head * DhN;
  const float* kr = row + Dn + head * DhN;
  const float* vr = row + 2 * Dn + head * DhN;

  float sq = 0.0f, sk = 0.0f;
#pragma unroll
  for (int j = 0; j < 4; ++j) {
    float a = qr[dl0 + j]; sq += a * a;
    float e = kr[dl0 + j]; sk += e * e;
  }
#pragma unroll
  for (int m = 8; m >= 1; m >>= 1) {
    sq += __shfl_xor(sq, m, 32);
    sk += __shfl_xor(sk, m, 32);
  }
  float rq = rsqrtf(sq * (1.0f / DhN) + RMS_EPS);
  float rk = rsqrtf(sk * (1.0f / DhN) + RMS_EPS);
  float p = (float)pos[t];
  size_t ob = (((size_t)(b * Hn + head)) * Tn + t) * DhN;

#pragma unroll
  for (int j = 0; j < 4; ++j) {
    int dl = dl0 + j;
    int dr = dl & 31;
    float fr = __powf(10000.0f, -(float)dr * (1.0f / 32.0f));
    float ang = p * fr;
    float cs = __cosf(ang), sn = __sinf(ang);
    int dp = (dl < 32) ? (dl + 32) : (dl - 32);
    float sgn = (dl < 32) ? -1.0f : 1.0f;
    float nq  = qr[dl] * rq * lnq[dl];
    float nqp = qr[dp] * rq * lnq[dp];
    float nk  = kr[dl] * rk * lnk[dl];
    float nkp = kr[dp] * rk * lnk[dp];
    qo[ob + dl] = f2bf(nq * cs + sgn * nqp * sn);
    ko[ob + dl] = f2bf(nk * cs + sgn * nkp * sn);
    vo[ob + dl] = f2bf(vr[dl]);
  }
}

// ---------------------------------------------------------------------------
// Flash attention (bf16 WMMA, f32 online softmax)
// grid (T/64, H, B); 128 threads = 4 waves; wave owns 16 q rows; KV block 64
// K tile async row-major copy; V tile staged transposed.
// ---------------------------------------------------------------------------
__global__ __launch_bounds__(128) void k_flash(const unsigned short* __restrict__ Q,
                                               const unsigned short* __restrict__ K,
                                               const unsigned short* __restrict__ V,
                                               unsigned short* __restrict__ O) {
  __shared__ unsigned short sK[64 * 64];        // [kv][d]
  __shared__ unsigned short sVt[64 * 64];       // [d][kv]  (transposed)
  __shared__ unsigned short sP[4 * 16 * 64];    // per-wave P staging
  int tid = threadIdx.x, lane = tid & 31, wid = tid >> 5;
  int h = blockIdx.y, b = blockIdx.z;
  int qbase = blockIdx.x * 64 + wid * 16;
  int ml = lane & 15, kb = (lane >> 4) * 8, mb = (lane < 16) ? 0 : 8;
  size_t headoff = ((size_t)(b * Hn + h)) * Tn * DhN;

  // Q fragments (held in registers for the whole kernel), b128 global loads
  bf16x16 qf[2];
  const unsigned short* qrow = Q + headoff + (size_t)(qbase + ml) * DhN;
#pragma unroll
  for (int f = 0; f < 2; ++f)
    qf[f] = mk_frag(qrow + f * 32 + kb, qrow + f * 32 + kb + 16);

  f32x8 o[4];
  float mOld[8], lS[8];
#pragma unroll
  for (int dt = 0; dt < 4; ++dt) o[dt] = zero8();
#pragma unroll
  for (int v = 0; v < 8; ++v) { mOld[v] = -1e30f; lS[v] = 0.0f; }

  const unsigned short* Kg = K + headoff;
  const unsigned int* Vu = (const unsigned int*)(V + headoff);

  for (int j0 = 0; j0 < Tn; j0 += 64) {
    // K tile: contiguous async b128 copy (512 x 16B)
#pragma unroll
    for (int j = 0; j < 4; ++j) {
      int idx = tid + j * 128;
      cp16(&Kg[(size_t)j0 * 64 + idx * 8], &sK[idx * 8]);
    }
    // V tile: transpose [kv][d] -> [d][kv]
#pragma unroll
    for (int u = 0; u < 2; ++u) {
      int np = tid & 31;                        // d-pair
      int kq = (tid >> 5) + u * 4;              // kv-octet 0..7
      unsigned int in[8];
#pragma unroll
      for (int i = 0; i < 8; ++i)
        in[i] = Vu[(size_t)(j0 + kq * 8 + i) * 32 + np];
      uint4v ta, tb;
#pragma unroll
      for (int j2 = 0; j2 < 4; ++j2) {
        ta[j2] = lo_pair(in[2 * j2], in[2 * j2 + 1]);
        tb[j2] = hi_pair(in[2 * j2], in[2 * j2 + 1]);
      }
      *(uint4v*)&sVt[(np * 2 + 0) * 64 + kq * 8] = ta;
      *(uint4v*)&sVt[(np * 2 + 1) * 64 + kq * 8] = tb;
    }
    cp16_commit();
    __syncthreads();

    // S = (Q @ K^T) * 1/sqrt(Dh)
    f32x8 s[4];
#pragma unroll
    for (int nt = 0; nt < 4; ++nt) {
      s[nt] = zero8();
      int n = nt * 16 + ml;
#pragma unroll
      for (int f = 0; f < 2; ++f) {
        bf16x16 bfr = mk_frag(&sK[n * 64 + f * 32 + kb], &sK[n * 64 + f * 32 + kb + 16]);
        s[nt] = __builtin_amdgcn_wmma_f32_16x16x32_bf16(
            false, qf[f], false, bfr, (short)0, s[nt], false, false);
      }
      s[nt] = s[nt] * 0.125f;
    }

    // online softmax
    float mNew[8], alpha[8], rs[8];
#pragma unroll
    for (int v = 0; v < 8; ++v) {
      float tmx = fmaxf(fmaxf(s[0][v], s[1][v]), fmaxf(s[2][v], s[3][v]));
#pragma unroll
      for (int msk = 8; msk >= 1; msk >>= 1) tmx = fmaxf(tmx, __shfl_xor(tmx, msk, 32));
      mNew[v] = fmaxf(mOld[v], tmx);
      alpha[v] = __expf(mOld[v] - mNew[v]);
      mOld[v] = mNew[v];
      rs[v] = 0.0f;
    }
#pragma unroll
    for (int nt = 0; nt < 4; ++nt)
#pragma unroll
      for (int v = 0; v < 8; ++v) {
        float pe = __expf(s[nt][v] - mNew[v]);
        s[nt][v] = pe;
        rs[v] += pe;
      }
#pragma unroll
    for (int v = 0; v < 8; ++v) {
      float r = rs[v];
#pragma unroll
      for (int msk = 8; msk >= 1; msk >>= 1) r += __shfl_xor(r, msk, 32);
      lS[v] = lS[v] * alpha[v] + r;
    }
#pragma unroll
    for (int dt = 0; dt < 4; ++dt)
#pragma unroll
      for (int v = 0; v < 8; ++v) o[dt][v] *= alpha[v];

    // stage P (C layout) -> LDS -> re-read in A layout
#pragma unroll
    for (int nt = 0; nt < 4; ++nt)
#pragma unroll
      for (int v = 0; v < 8; ++v)
        sP[wid * 1024 + (mb + v) * 64 + nt * 16 + ml] = f2bf(s[nt][v]);
    __syncthreads();

    // O += P @ V
#pragma unroll
    for (int kc = 0; kc < 2; ++kc) {
      bf16x16 pa = mk_frag(&sP[wid * 1024 + ml * 64 + kc * 32 + kb],
                           &sP[wid * 1024 + ml * 64 + kc * 32 + kb + 16]);
#pragma unroll
      for (int dt = 0; dt < 4; ++dt) {
        int n = dt * 16 + ml;
        bf16x16 bv = mk_frag(&sVt[n * 64 + kc * 32 + kb], &sVt[n * 64 + kc * 32 + kb + 16]);
        o[dt] = __builtin_amdgcn_wmma_f32_16x16x32_bf16(
            false, pa, false, bv, (short)0, o[dt], false, false);
      }
    }
    __syncthreads();
  }

  // O /= l ; scatter to attn_out bf16 [M, D] with heads re-concatenated
  size_t orow = (size_t)(b * Tn);
#pragma unroll
  for (int dt = 0; dt < 4; ++dt)
#pragma unroll
    for (int v = 0; v < 8; ++v) {
      float val = o[dt][v] / lS[v];
      int trow = qbase + mb + v;
      O[(orow + trow) * Dn + h * DhN + dt * 16 + ml] = f2bf(val);
    }
}

// ---------------------------------------------------------------------------
// x2 = x + gate_msa*proj_out ; h = bf16( RMS(x2)*(1+scale_mlp)+shift_mlp )
// ---------------------------------------------------------------------------
__global__ __launch_bounds__(256) void k_resid1(const float* __restrict__ x,
                                                const float* __restrict__ pj,
                                                const float* __restrict__ mods,
                                                float* __restrict__ x2,
                                                unsigned short* __restrict__ hb) {
  int row = blockIdx.x;
  int b = row >> 11;
  int tid = threadIdx.x;
  const float* mo = mods + b * (6 * Dn);
  float v[4];
  float ss = 0.0f;
#pragma unroll
  for (int j = 0; j < 4; ++j) {
    int d = tid * 4 + j;
    size_t idx = (size_t)row * Dn + d;
    float val = x[idx] + mo[2 * Dn + d] * pj[idx];   // gate_msa (bias already in pj)
    v[j] = val;
    ss += val * val;
    x2[idx] = val;
  }
  __shared__ float red[256];
  red[tid] = ss;
  __syncthreads();
  for (int s = 128; s > 0; s >>= 1) {
    if (tid < s) red[tid] += red[tid + s];
    __syncthreads();
  }
  float rs = rsqrtf(red[0] * (1.0f / Dn) + RMS_EPS);
  unsigned int pk[2];
#pragma unroll
  for (int j = 0; j < 4; ++j) {
    int d = tid * 4 + j;
    float hh = v[j] * rs * (1.0f + mo[4 * Dn + d]) + mo[3 * Dn + d];
    unsigned short hv = f2bf(hh);
    if (j & 1) pk[j >> 1] |= ((unsigned int)hv << 16);
    else       pk[j >> 1]  = hv;
  }
  *(unsigned int*)&hb[(size_t)row * Dn + tid * 4]     = pk[0];
  *(unsigned int*)&hb[(size_t)row * Dn + tid * 4 + 2] = pk[1];
}

// ---------------------------------------------------------------------------
// out = x2 + gate_mlp * mlp_out
// ---------------------------------------------------------------------------
__global__ __launch_bounds__(256) void k_final(const float* __restrict__ x2,
                                               const float* __restrict__ mlp,
                                               const float* __restrict__ mods,
                                               float* __restrict__ out) {
  int i0 = (blockIdx.x * 256 + threadIdx.x) * 4;
  int row = i0 >> 10;
  int b = row >> 11;
  int d = i0 & (Dn - 1);
  const float* mo = mods + b * (6 * Dn) + 5 * Dn;
#pragma unroll
  for (int j = 0; j < 4; ++j) out[i0 + j] = x2[i0 + j] + mo[d + j] * mlp[i0 + j];
}

// ---------------------------------------------------------------------------
extern "C" void kernel_launch(void* const* d_in, const int* in_sizes, int n_in,
                              void* d_out, int out_size, void* d_ws, size_t ws_size,
                              hipStream_t stream) {
  (void)in_sizes; (void)n_in; (void)out_size; (void)ws_size;
  const float* x       = (const float*)d_in[0];
  const float* c       = (const float*)d_in[1];
  const int*   pos     = (const int*)d_in[2];
  const float* adaln_w = (const float*)d_in[3];
  const float* adaln_b = (const float*)d_in[4];
  const float* qkv_w   = (const float*)d_in[5];
  const float* lnq_w   = (const float*)d_in[6];
  const float* lnk_w   = (const float*)d_in[7];
  const float* proj_w  = (const float*)d_in[8];
  const float* proj_b  = (const float*)d_in[9];
  const float* w1_w    = (const float*)d_in[10];
  const float* w3_w    = (const float*)d_in[11];
  const float* w2_w    = (const float*)d_in[12];
  float* out = (float*)d_out;

  char* ws = (char*)d_ws;
  size_t off = 0;
  auto alloc = [&](size_t bytes) -> char* {
    char* p = ws + off;
    off += (bytes + 255) & ~(size_t)255;
    return p;
  };
  float*          csilu  = (float*)alloc((size_t)Bn * Dn * 4);
  float*          mods   = (float*)alloc((size_t)Bn * 6 * Dn * 4);
  unsigned short* xmod   = (unsigned short*)alloc((size_t)Mn * Dn * 2);
  unsigned short* qkvw   = (unsigned short*)alloc((size_t)Dn * 3 * Dn * 2);
  unsigned short* projw  = (unsigned short*)alloc((size_t)Dn * Dn * 2);
  unsigned short* w1b    = (unsigned short*)alloc((size_t)Dn * HmN * 2);
  unsigned short* w3b    = (unsigned short*)alloc((size_t)Dn * HmN * 2);
  unsigned short* w2b    = (unsigned short*)alloc((size_t)HmN * Dn * 2);
  float*          qkvf   = (float*)alloc((size_t)Mn * 3 * Dn * 4);
  unsigned short* qb     = (unsigned short*)alloc((size_t)Mn * Dn * 2);
  unsigned short* kbuf   = (unsigned short*)alloc((size_t)Mn * Dn * 2);
  unsigned short* vb     = (unsigned short*)alloc((size_t)Mn * Dn * 2);
  unsigned short* attnb  = (unsigned short*)alloc((size_t)Mn * Dn * 2);
  float*          pjout  = (float*)alloc((size_t)Mn * Dn * 4);
  float*          x2     = (float*)alloc((size_t)Mn * Dn * 4);
  unsigned short* hb     = (unsigned short*)alloc((size_t)Mn * Dn * 2);
  unsigned short* gb     = (unsigned short*)alloc((size_t)Mn * HmN * 2);
  float*          mlpout = (float*)alloc((size_t)Mn * Dn * 4);

  auto cvt = [&](const float* s, unsigned short* dst, int n) {
    k_cvt<<<(n / 4 + 255) / 256, 256, 0, stream>>>(s, dst, n);
  };
  cvt(qkv_w, qkvw, Dn * 3 * Dn);
  cvt(proj_w, projw, Dn * Dn);
  cvt(w1_w, w1b, Dn * HmN);
  cvt(w3_w, w3b, Dn * HmN);
  cvt(w2_w, w2b, HmN * Dn);

  k_silu<<<(Bn * Dn + 255) / 256, 256, 0, stream>>>(c, csilu, Bn * Dn);
  k_adaln<<<(Bn * 6 * Dn) / 256, 256, 0, stream>>>(csilu, adaln_w, adaln_b, mods);
  k_modrms<<<Mn, 256, 0, stream>>>(x, mods, xmod);
  k_gemm<3 * Dn, Dn><<<dim3((3 * Dn) / 128, Mn / 128), 256, 0, stream>>>(xmod, qkvw, qkvf, nullptr);
  k_qkvprep<<<dim3(Tn, Bn), 256, 0, stream>>>(qkvf, lnq_w, lnk_w, pos, qb, kbuf, vb);
  k_flash<<<dim3(Tn / 64, Hn, Bn), 128, 0, stream>>>(qb, kbuf, vb, attnb);
  k_gemm<Dn, Dn><<<dim3(Dn / 128, Mn / 128), 256, 0, stream>>>(attnb, projw, pjout, proj_b);
  k_resid1<<<Mn, 256, 0, stream>>>(x, pjout, mods, x2, hb);
  k_gemm_dual<HmN, Dn><<<dim3(HmN / 64, Mn / 64), 128, 0, stream>>>(hb, w1b, w3b, gb);
  k_gemm<Dn, HmN><<<dim3(Dn / 128, Mn / 128), 256, 0, stream>>>(gb, w2b, mlpout, nullptr);
  k_final<<<Mn, 256, 0, stream>>>(x2, mlpout, mods, out);
}
